// locationAttention_2284922602062
// MI455X (gfx1250) — matrix operand order repaired and checked by reading, not compiled, not run on hardware
//
#include <hip/hip_runtime.h>
#include <hip/hip_bf16.h>

typedef __attribute__((ext_vector_type(16))) _Float16 v16h;
typedef __attribute__((ext_vector_type(8)))  float    v8f;

constexpr int LL = 2, BB = 32, TSZ = 2048, HH = 512, KK = 64, RR = 7;
constexpr int TTILE = 128;           // t rows per block (8 waves x 16)
constexpr int NT = HH / 16;          // 32 n-tiles over H

__device__ __forceinline__ float fast_tanh(float x) {
#if __has_builtin(__builtin_amdgcn_tanhf)
    return __builtin_amdgcn_tanhf(x);        // v_tanh_f32 (single TRANS op)
#else
    // tanh(x) = sign(x) * (1 - 2/(exp(2|x|)+1)), with fast rcp (no IEEE div)
    float ax = __builtin_fabsf(x);
    float e  = __expf(2.0f * ax);            // v_exp_f32 path
    float r  = __builtin_amdgcn_rcpf(e + 1.0f);   // v_rcp_f32
    float t  = __builtin_fmaf(-2.0f, r, 1.0f);
    return __builtin_copysignf(t, x);
#endif
}

// ---------------------------------------------------------------------------
// Kernel 1: prep
//  blocks [0,BB):   pre[b,h] = mean_L(hidden) @ hp_w.T + hp_b + pa_b
//  blocks [BB,BB+8): repack pa_w (H,K) f32 -> f16 in WMMA B-fragment order:
//     frag_pa[((nt*2+ks)*32 + lane)*16 + i] =
//         pa_w[h = nt*16+(lane&15)][k = ks*32 + (lane&16 ? 8:0) + (i<8 ? i : i+8)]
// ---------------------------------------------------------------------------
__global__ __launch_bounds__(256) void prep_kernel(
    const float* __restrict__ hidden, const float* __restrict__ hp_w,
    const float* __restrict__ hp_b,   const float* __restrict__ pa_w,
    const float* __restrict__ pa_b,
    float* __restrict__ pre, _Float16* __restrict__ frag_pa)
{
    const int tid = threadIdx.x;
    const int blk = blockIdx.x;
    if (blk < BB) {
        __shared__ float hmean[HH];
        const int b = blk;
        for (int j = tid; j < HH; j += 256)
            hmean[j] = 0.5f * (hidden[b * HH + j] + hidden[BB * HH + b * HH + j]);
        __syncthreads();
        for (int h = tid; h < HH; h += 256) {
            const float4* wrow = (const float4*)(hp_w + (size_t)h * HH);
            const float4* hm4  = (const float4*)hmean;
            float acc = 0.0f;
            #pragma unroll 4
            for (int j = 0; j < HH / 4; ++j) {
                float4 w4 = wrow[j];
                float4 m4 = hm4[j];
                acc += w4.x * m4.x + w4.y * m4.y + w4.z * m4.z + w4.w * m4.w;
            }
            pre[b * HH + h] = acc + hp_b[h] + pa_b[h];
        }
    } else {
        const int bb = blk - BB;                       // 0..7
        for (int e = bb * 8192 + tid; e < (bb + 1) * 8192; e += 256) {
            const int i    = e & 15;
            const int lane = (e >> 4) & 31;
            const int ksnt = e >> 9;
            const int ks   = ksnt & 1;
            const int nt   = ksnt >> 1;
            const int h    = nt * 16 + (lane & 15);
            const int k    = ks * 32 + ((lane & 16) ? 8 : 0) + ((i < 8) ? i : i + 8);
            frag_pa[e] = (_Float16)pa_w[h * KK + k];
        }
    }
}

// ---------------------------------------------------------------------------
// Kernel 2: fused conv -> WMMA GEMM -> tanh -> out_w dot  => energy (B,T)
// grid (T/TTILE, B), 256 threads = 8 waves, each wave a 16-row t tile.
// ---------------------------------------------------------------------------
__global__ __launch_bounds__(256) void energy_kernel(
    const float* __restrict__ enc,      // (T,B,H)
    const float* __restrict__ prev,     // (B,T)
    const float* __restrict__ conv_w,   // (K,1,R)
    const float* __restrict__ conv_b,   // (K)
    const float* __restrict__ out_w,    // (1,H)
    const float* __restrict__ out_b,    // (1)
    const float* __restrict__ pre,      // (B,H)
    const _Float16* __restrict__ frag_pa,
    float* __restrict__ energy)         // (B,T)
{
    __shared__ float s_w[KK * RR];
    __shared__ float s_b[KK];
    const int tid = threadIdx.x;
    for (int j = tid; j < KK * RR; j += 256) s_w[j] = conv_w[j];
    for (int j = tid; j < KK; j += 256)      s_b[j] = conv_b[j];
    __syncthreads();

    const int b      = blockIdx.y;
    const int tbase  = blockIdx.x * TTILE;
    const int wave   = tid >> 5;
    const int lane   = tid & 31;
    const int lane16 = lane & 15;
    const bool hi    = (lane & 16) != 0;
    const int tw     = tbase + wave * 16;

    // --- conv window for this lane's A-matrix row (m = lane16) ---
    const int tm = tw + lane16;
    float p[RR];
    #pragma unroll
    for (int r = 0; r < RR; ++r) {
        const int idx = tm - 3 + r;
        p[r] = (idx >= 0 && idx < TSZ) ? prev[b * TSZ + idx] : 0.0f;
    }

    // --- build A fragments in ISA 16-bit A 16x32 lane layout ---
    // lanes 0-15 : element i -> k = (i<8 ? i   : i+8 )   (K 0..7, 16..23)
    // lanes 16-31: element i -> k = (i<8 ? i+8 : i+16)   (K 8..15, 24..31)
    v16h a0, a1;
    #pragma unroll
    for (int i = 0; i < 16; ++i) {
        const int koff = hi ? ((i < 8) ? i + 8 : i + 16)
                            : ((i < 8) ? i     : i + 8 );
        {
            const int k = koff;
            float y = s_b[k];
            #pragma unroll
            for (int r = 0; r < RR; ++r) y += s_w[k * RR + r] * p[r];
            a0[i] = (_Float16)y;
        }
        {
            const int k = 32 + koff;
            float y = s_b[k];
            #pragma unroll
            for (int r = 0; r < RR; ++r) y += s_w[k * RR + r] * p[r];
            a1[i] = (_Float16)y;
        }
    }

    float eacc[8];
    #pragma unroll
    for (int r = 0; r < 8; ++r) eacc[r] = 0.0f;

    // C layout: VGPR r, lanes 0-15  -> (M=r,   N=lane16)
    //                   lanes 16-31 -> (M=r+8, N=lane16)
    const int trow0 = tw + (hi ? 8 : 0);
    const float* encp = enc + (size_t)trow0 * BB * HH + (size_t)b * HH + lane16;

    #pragma unroll 2
    for (int nt = 0; nt < NT; ++nt) {
        const int h = nt * 16 + lane16;

        // B fragments: contiguous 32B per lane, prepacked in prep_kernel
        union { uint4 u[2]; v16h h16; } ub0, ub1;
        const uint4* bp0 = (const uint4*)(frag_pa + ((size_t)(nt * 2    ) * 32 + lane) * 16);
        const uint4* bp1 = (const uint4*)(frag_pa + ((size_t)(nt * 2 + 1) * 32 + lane) * 16);
        ub0.u[0] = bp0[0]; ub0.u[1] = bp0[1];
        ub1.u[0] = bp1[0]; ub1.u[1] = bp1[1];

        v8f c = {};
        c = __builtin_amdgcn_wmma_f32_16x16x32_f16(false, a0, false, ub0.h16,
                                                   (short)0, c, false, false);
        c = __builtin_amdgcn_wmma_f32_16x16x32_f16(false, a1, false, ub1.h16,
                                                   (short)0, c, false, false);

        const float wv = out_w[h];
        const float pv = pre[b * HH + h];
        #pragma unroll
        for (int r = 0; r < 8; ++r) {
            const float ev = c[r] + encp[(size_t)r * BB * HH] + pv;
            eacc[r] += fast_tanh(ev) * wv;
        }
        encp += 16;   // h advances by 16 floats
    }

    const float ob = *out_b;
    #pragma unroll
    for (int r = 0; r < 8; ++r) {
        float red = eacc[r];
        #pragma unroll
        for (int off = 8; off >= 1; off >>= 1)
            red += __shfl_xor(red, off, 32);
        if (lane16 == 0)                       // lane 0 and lane 16 each write
            energy[b * TSZ + trow0 + r] = red + ob;
    }
}

// ---------------------------------------------------------------------------
// Kernel 3: masked softmax over T per batch row
// ---------------------------------------------------------------------------
__global__ __launch_bounds__(256) void softmax_kernel(
    const float* __restrict__ energy, const int* __restrict__ enc_len,
    float* __restrict__ out)
{
    __shared__ float red[256];
    const int b   = blockIdx.x;
    const int tid = threadIdx.x;
    const int len = enc_len[b];

    float vals[TSZ / 256];
    float mx = -3.0e38f;
    #pragma unroll
    for (int j = 0; j < TSZ / 256; ++j) {
        const int t = tid + j * 256;
        const float v = (t < len) ? energy[b * TSZ + t] : -1.0e30f;
        vals[j] = v;
        mx = fmaxf(mx, v);
    }
    red[tid] = mx;
    __syncthreads();
    for (int s = 128; s > 0; s >>= 1) {
        if (tid < s) red[tid] = fmaxf(red[tid], red[tid + s]);
        __syncthreads();
    }
    mx = red[0];
    __syncthreads();

    float sum = 0.0f;
    #pragma unroll
    for (int j = 0; j < TSZ / 256; ++j) {
        const float e = __expf(vals[j] - mx);
        vals[j] = e;
        sum += e;
    }
    red[tid] = sum;
    __syncthreads();
    for (int s = 128; s > 0; s >>= 1) {
        if (tid < s) red[tid] += red[tid + s];
        __syncthreads();
    }
    const float inv = 1.0f / red[0];
    #pragma unroll
    for (int j = 0; j < TSZ / 256; ++j) {
        const int t = tid + j * 256;
        out[b * TSZ + t] = (t < len) ? vals[j] * inv : 0.0f;
    }
}

// ---------------------------------------------------------------------------
extern "C" void kernel_launch(void* const* d_in, const int* in_sizes, int n_in,
                              void* d_out, int out_size, void* d_ws, size_t ws_size,
                              hipStream_t stream)
{
    const float* hidden  = (const float*)d_in[0];
    const float* enc     = (const float*)d_in[1];
    const int*   enc_len = (const int*)  d_in[2];
    const float* prev    = (const float*)d_in[3];
    const float* conv_w  = (const float*)d_in[4];
    const float* conv_b  = (const float*)d_in[5];
    const float* hp_w    = (const float*)d_in[6];
    const float* hp_b    = (const float*)d_in[7];
    const float* pa_w    = (const float*)d_in[8];
    const float* pa_b    = (const float*)d_in[9];
    const float* out_w   = (const float*)d_in[10];
    const float* out_b   = (const float*)d_in[11];

    char* ws = (char*)d_ws;
    float*    pre     = (float*)ws;                   // B*H f32      = 64 KB
    _Float16* frag_pa = (_Float16*)(ws + 65536);      // 32768 f16    = 64 KB
    float*    energy  = (float*)(ws + 131072);        // B*T f32      = 256 KB

    prep_kernel<<<BB + 8, 256, 0, stream>>>(hidden, hp_w, hp_b, pa_w, pa_b,
                                            pre, frag_pa);
    energy_kernel<<<dim3(TSZ / TTILE, BB), 256, 0, stream>>>(
        enc, prev, conv_w, conv_b, out_w, out_b, pre, frag_pa, energy);
    softmax_kernel<<<BB, 256, 0, stream>>>(energy, enc_len, (float*)d_out);
}
